// SelectiveSSM_43877385896137
// MI455X (gfx1250) — matrix operand order repaired and checked by reading, not compiled, not run on hardware
//
#include <hip/hip_runtime.h>

// ---------------------------------------------------------------------------
// SelectiveSSM for MI455X (gfx1250, wave32).
// GEMMs: v_wmma_f32_16x16x32_bf16 with LDS-staged, double-buffered B tiles
// (block = 128x128 tile, 8 waves, cooperative fill) and software pipelining.
// The L-sequential scan is parallelized as a 3-pass chunked scan.
// ---------------------------------------------------------------------------

typedef __bf16 bf16;
typedef __attribute__((ext_vector_type(8)))  __bf16 v8bf;
typedef __attribute__((ext_vector_type(16))) __bf16 v16bf;
typedef __attribute__((ext_vector_type(8)))  float  v8f;

#define D_MODEL 768
#define D_STATE 16
#define BATCH   2
#define SEQLEN  2048
#define ROWS    (BATCH * SEQLEN)      // 4096 token rows
#define KBIG    (D_MODEL * D_STATE)   // 12288
#define CHUNK   128
#define NCHUNK  (SEQLEN / CHUNK)      // 16

// ---------------------------------------------------------------------------
// WMMA helpers (this code's contiguous k order, used for both A and B; not the ISA table order).
// A (16x32 bf16, MxK): lane<16 -> row=lane, K=[k0+0..15];
//                      lane>=16 -> row=lane-16, K=[k0+16..31]
// ---------------------------------------------------------------------------
__device__ __forceinline__ v16bf load_tileA(const bf16* __restrict__ row_base,
                                            int k0, int lane) {
  const int half = lane >> 4;
  v8bf c0 = *(const v8bf*)(row_base + k0 + 16 * half);
  v8bf c1 = *(const v8bf*)(row_base + k0 + 8 + 16 * half);
  v16bf a;
#pragma unroll
  for (int i = 0; i < 8; ++i) { a[i] = c0[i]; a[i + 8] = c1[i]; }
  return a;
}

// B fragment (32x16 KxN): lane holds column n=lane&15, K-halves split at lane 16;
// each lane's fragment is 16 contiguous bf16 from weight row (colBase+n).
__device__ __forceinline__ v8f wmma_bf16(v16bf a, v16bf b, v8f c) {
  return __builtin_amdgcn_wmma_f32_16x16x32_bf16(
      /*neg_a=*/false, a, /*neg_b=*/false, b,
      /*c_mod=*/(short)0, c, /*reuse_a=*/false, /*reuse_b=*/false);
}

__device__ __forceinline__ float softplusf(float x) {
  return x > 20.f ? x : __logf(1.f + __expf(x));
}

// ---------------------------------------------------------------------------
// Elementwise helpers
// ---------------------------------------------------------------------------
__global__ void cvt_f32_bf16_kernel(const float* __restrict__ in,
                                    bf16* __restrict__ out, int n) {
  int i = blockIdx.x * blockDim.x + threadIdx.x;
  if (i < n) out[i] = (bf16)in[i];
}

__global__ void lam_kernel(const float* __restrict__ log_lambda,
                           float* __restrict__ lam, int n) {
  int i = blockIdx.x * blockDim.x + threadIdx.x;
  if (i < n) lam[i] = softplusf(log_lambda[i]);
}

// ---------------------------------------------------------------------------
// Tiled GEMM: dst[r, c] = f(A[r, :K] . W[c, :K] + bias[c])
//   block = 128 rows x 128 cols, 8 waves; wave = 16 rows x 128 cols (8 accs).
//   B staged through LDS in fragment order, double-buffered, one barrier/stage.
//   KTOT is both the K extent and the row stride of A and W (true here).
// ---------------------------------------------------------------------------
template <int KTOT, bool SOFTPLUS>
__global__ __launch_bounds__(256)
void gemm128_kernel(const bf16* __restrict__ A, const bf16* __restrict__ W,
                    const float* __restrict__ bias, float* __restrict__ dst) {
  __shared__ __align__(16) bf16 sB[2 * 4096];   // 2 x 8KB staged B fragments

  const int tid  = threadIdx.x;
  const int lane = tid & 31;
  const int wave = tid >> 5;
  const int half = lane >> 4;
  const int n    = lane & 15;

  const int rowBlock = blockIdx.x / 6;          // 32 row-blocks of 128 rows
  const int colBase  = (blockIdx.x % 6) * 128;  // 6 strips of 128 cols
  const int rowTile  = rowBlock * 8 + wave;     // this wave's 16-row tile
  const bf16* arow = A + (size_t)(rowTile * 16 + n) * KTOT;

  // Cooperative fill: thread tid stages the 32B fragment (tile ft, lane fl).
  const int ft = tid >> 5;                      // B col-tile 0..7
  const int fl = tid & 31;
  const bf16* fsrc =
      W + (size_t)(colBase + ft * 16 + (fl & 15)) * KTOT + 16 * (fl >> 4);
  bf16* sdst = sB + ft * 512 + fl * 16;         // fragment-order slot (32B)

  // Prologue: stage k0 = 0 into buffer 0.
  {
    v8bf f0 = *(const v8bf*)(fsrc);
    v8bf f1 = *(const v8bf*)(fsrc + 8);
    *(v8bf*)(sdst)     = f0;
    *(v8bf*)(sdst + 8) = f1;
  }
  __syncthreads();

  v8f zero = {};
  v8f acc[8];
#pragma unroll
  for (int t = 0; t < 8; ++t) acc[t] = zero;

  constexpr int NIT = KTOT / 32;
  for (int it = 0; it < NIT; ++it) {
    const int k0  = it * 32;
    const int cur = it & 1;

    // (a) A fragment first: its wait leaves next-stage loads outstanding.
    v16bf a = load_tileA(arow, k0, lane);

    // (b) Next-stage B global loads — complete under the WMMAs below.
    v8bf f0, f1;
    const bool haveNext = (it + 1 < NIT);
    if (haveNext) {
      __builtin_prefetch(fsrc + k0 + 96, 0, 1);      // global_prefetch_b8
      f0 = *(const v8bf*)(fsrc + k0 + 32);
      f1 = *(const v8bf*)(fsrc + k0 + 40);
    }

    // (c) Consume current LDS buffer: 8 fragments -> 8 WMMAs sharing one A.
    const bf16* sb = sB + cur * 4096;
#pragma unroll
    for (int t = 0; t < 8; ++t) {
      const v8bf* p = (const v8bf*)(sb + t * 512 + lane * 16);
      v8bf c0 = p[0];
      v8bf c1 = p[1];
      v16bf b;
#pragma unroll
      for (int i = 0; i < 8; ++i) { b[i] = c0[i]; b[i + 8] = c1[i]; }
      acc[t] = wmma_bf16(a, b, acc[t]);
    }

    // (d) Commit next stage to the other buffer, (e) one barrier per stage.
    if (haveNext) {
      bf16* d = sB + (cur ^ 1) * 4096 + ft * 512 + fl * 16;
      *(v8bf*)(d)     = f0;
      *(v8bf*)(d + 8) = f1;
    }
    __syncthreads();
  }

  // Epilogue: C layout — VGPR i holds M=i (lanes 0-15) / M=i+8 (lanes 16-31).
  float bv[8];
#pragma unroll
  for (int t = 0; t < 8; ++t) bv[t] = bias[colBase + t * 16 + n];
#pragma unroll
  for (int i = 0; i < 8; ++i) {
    const int r = rowTile * 16 + i + 8 * half;
    float* orow = dst + (size_t)r * D_MODEL + colBase;
#pragma unroll
    for (int t = 0; t < 8; ++t) {
      float v = acc[t][i] + bv[t];
      if (SOFTPLUS) v = softplusf(v);
      orow[t * 16 + n] = v;
    }
  }
}

// ---------------------------------------------------------------------------
// Bt = x @ WB^T, Ct = x @ WC^T   [ROWS x 16] each, K = 768. One wave per
// 16-row tile computes both 16x16 outputs sharing the A tile. (0.2 GF: tiny.)
// ---------------------------------------------------------------------------
__device__ __forceinline__ v16bf load_tileB_g(const bf16* __restrict__ w, int ld,
                                              int k0, int lane) {
  const int nn = lane & 15;
  const int half = lane >> 4;
  const bf16* p = w + (size_t)nn * ld + k0 + 16 * half;
  v8bf c0 = *(const v8bf*)(p);
  v8bf c1 = *(const v8bf*)(p + 8);
  v16bf b;
#pragma unroll
  for (int i = 0; i < 8; ++i) { b[i] = c0[i]; b[i + 8] = c1[i]; }
  return b;
}

__global__ __launch_bounds__(256)
void btct_kernel(const bf16* __restrict__ X, const bf16* __restrict__ WB,
                 const bf16* __restrict__ WC, float* __restrict__ Bt,
                 float* __restrict__ Ct) {
  const int lane = threadIdx.x & 31;
  const int rowTile = blockIdx.x * 8 + (threadIdx.x >> 5);   // 0..255
  const int half = lane >> 4;
  const bf16* arow = X + (size_t)(rowTile * 16 + (lane & 15)) * D_MODEL;

  v8f aB = {}, aC = {};
  for (int k0 = 0; k0 < D_MODEL; k0 += 32) {
    v16bf a  = load_tileA(arow, k0, lane);
    v16bf bb = load_tileB_g(WB, D_MODEL, k0, lane);
    v16bf bc = load_tileB_g(WC, D_MODEL, k0, lane);
    aB = wmma_bf16(a, bb, aB);
    aC = wmma_bf16(a, bc, aC);
  }
  const int n = lane & 15;
#pragma unroll
  for (int i = 0; i < 8; ++i) {
    const int r = rowTile * 16 + i + 8 * half;
    Bt[r * D_STATE + n] = aB[i];
    Ct[r * D_STATE + n] = aC[i];
  }
}

// ---------------------------------------------------------------------------
// Chunked selective scan. Thread <-> (b, d, chunk); each thread owns 16 states.
//   a[l,d,n] = exp(-delta[l,d] * lam[d,n]);  bterm = delta[l,d] * Bt[l,n]
// Pass 1: per-chunk transition  h_out = P*h_in + Q  (P=prod a, Q=local scan)
// ---------------------------------------------------------------------------
__global__ __launch_bounds__(256)
void scan_pass1_kernel(const float* __restrict__ delta,
                       const float* __restrict__ Bt,
                       const float* __restrict__ lam,
                       float* __restrict__ P, float* __restrict__ Q) {
  const int tid = blockIdx.x * blockDim.x + threadIdx.x;   // < 2*768*16
  const int d = tid % D_MODEL;
  const int c = (tid / D_MODEL) % NCHUNK;
  const int b = tid / (D_MODEL * NCHUNK);

  float lamv[D_STATE];
  const float4* lp = (const float4*)(lam + d * D_STATE);
#pragma unroll
  for (int q = 0; q < 4; ++q) {
    float4 v = lp[q];
    lamv[4*q] = v.x; lamv[4*q+1] = v.y; lamv[4*q+2] = v.z; lamv[4*q+3] = v.w;
  }
  float Pv[D_STATE], Qv[D_STATE];
#pragma unroll
  for (int nn = 0; nn < D_STATE; ++nn) { Pv[nn] = 1.f; Qv[nn] = 0.f; }

  const int l0 = c * CHUNK;
  for (int l = l0; l < l0 + CHUNK; ++l) {
    const int row = b * SEQLEN + l;
    const float dl = delta[(size_t)row * D_MODEL + d];
    const float4* bp = (const float4*)(Bt + row * D_STATE);
    float bt[D_STATE];
#pragma unroll
    for (int q = 0; q < 4; ++q) {
      float4 v = bp[q];
      bt[4*q] = v.x; bt[4*q+1] = v.y; bt[4*q+2] = v.z; bt[4*q+3] = v.w;
    }
#pragma unroll
    for (int nn = 0; nn < D_STATE; ++nn) {
      const float a = __expf(-dl * lamv[nn]);
      Pv[nn] *= a;
      Qv[nn] = fmaf(a, Qv[nn], dl * bt[nn]);
    }
  }
  const size_t base = (((size_t)b * D_MODEL + d) * NCHUNK + c) * D_STATE;
#pragma unroll
  for (int nn = 0; nn < D_STATE; ++nn) { P[base + nn] = Pv[nn]; Q[base + nn] = Qv[nn]; }
}

// Pass 2: combine 16 chunk transitions sequentially; record h at chunk starts.
__global__ __launch_bounds__(256)
void scan_pass2_kernel(const float* __restrict__ P, const float* __restrict__ Q,
                       float* __restrict__ Hs) {
  const int tid = blockIdx.x * blockDim.x + threadIdx.x;   // < 2*768
  const int d = tid % D_MODEL;
  const int b = tid / D_MODEL;
  float h[D_STATE];
#pragma unroll
  for (int nn = 0; nn < D_STATE; ++nn) h[nn] = 0.f;
  for (int c = 0; c < NCHUNK; ++c) {
    const size_t base = (((size_t)b * D_MODEL + d) * NCHUNK + c) * D_STATE;
#pragma unroll
    for (int nn = 0; nn < D_STATE; ++nn) {
      Hs[base + nn] = h[nn];
      h[nn] = fmaf(P[base + nn], h[nn], Q[base + nn]);
    }
  }
}

// Pass 3: replay each chunk from its true initial state; emit y = h*Ct as bf16
// into flat[row, d*16 + n] (the A operand of the output GEMM).
__global__ __launch_bounds__(256)
void scan_pass3_kernel(const float* __restrict__ delta,
                       const float* __restrict__ Bt,
                       const float* __restrict__ Ct,
                       const float* __restrict__ lam,
                       const float* __restrict__ Hs,
                       bf16* __restrict__ flat) {
  const int tid = blockIdx.x * blockDim.x + threadIdx.x;   // < 2*768*16
  const int d = tid % D_MODEL;
  const int c = (tid / D_MODEL) % NCHUNK;
  const int b = tid / (D_MODEL * NCHUNK);

  float lamv[D_STATE];
  const float4* lp = (const float4*)(lam + d * D_STATE);
#pragma unroll
  for (int q = 0; q < 4; ++q) {
    float4 v = lp[q];
    lamv[4*q] = v.x; lamv[4*q+1] = v.y; lamv[4*q+2] = v.z; lamv[4*q+3] = v.w;
  }
  float h[D_STATE];
  const size_t hbase = (((size_t)b * D_MODEL + d) * NCHUNK + c) * D_STATE;
#pragma unroll
  for (int nn = 0; nn < D_STATE; ++nn) h[nn] = Hs[hbase + nn];

  const int l0 = c * CHUNK;
  for (int l = l0; l < l0 + CHUNK; ++l) {
    const int row = b * SEQLEN + l;
    const float dl = delta[(size_t)row * D_MODEL + d];
    const float4* bp = (const float4*)(Bt + row * D_STATE);
    const float4* cp = (const float4*)(Ct + row * D_STATE);
    float bt[D_STATE], ct[D_STATE];
#pragma unroll
    for (int q = 0; q < 4; ++q) {
      float4 v = bp[q];
      bt[4*q] = v.x; bt[4*q+1] = v.y; bt[4*q+2] = v.z; bt[4*q+3] = v.w;
      float4 w = cp[q];
      ct[4*q] = w.x; ct[4*q+1] = w.y; ct[4*q+2] = w.z; ct[4*q+3] = w.w;
    }
    v8bf s0, s1;
#pragma unroll
    for (int nn = 0; nn < D_STATE; ++nn) {
      const float a = __expf(-dl * lamv[nn]);
      h[nn] = fmaf(a, h[nn], dl * bt[nn]);
      const float y = h[nn] * ct[nn];
      if (nn < 8) s0[nn] = (bf16)y; else s1[nn - 8] = (bf16)y;
    }
    bf16* fp = flat + (size_t)row * KBIG + d * D_STATE;
    *(v8bf*)(fp)     = s0;
    *(v8bf*)(fp + 8) = s1;
  }
}

// ---------------------------------------------------------------------------
// Host-side orchestration (graph-capture safe: only kernel launches on stream)
// ---------------------------------------------------------------------------
extern "C" void kernel_launch(void* const* d_in, const int* in_sizes, int n_in,
                              void* d_out, int out_size, void* d_ws, size_t ws_size,
                              hipStream_t stream) {
  const float* x    = (const float*)d_in[0];   // [2,2048,768]
  const float* llam = (const float*)d_in[1];   // [768,16]
  const float* Wd   = (const float*)d_in[2];   // [768,768]
  const float* bd   = (const float*)d_in[3];   // [768]
  const float* WB   = (const float*)d_in[4];   // [16,768]
  const float* WC   = (const float*)d_in[5];   // [16,768]
  const float* Wout = (const float*)d_in[6];   // [768,12288]
  const float* bout = (const float*)d_in[7];   // [768]
  float* out = (float*)d_out;

  char* ws = (char*)d_ws;
  size_t off = 0;
  auto alloc = [&](size_t bytes) -> void* {
    void* p = ws + off;
    off = (off + bytes + 255) & ~(size_t)255;
    return p;
  };
  bf16*  xb    = (bf16*)alloc((size_t)ROWS * D_MODEL * 2);        // 6.3 MB
  bf16*  Wdb   = (bf16*)alloc((size_t)D_MODEL * D_MODEL * 2);     // 1.2 MB
  bf16*  WBb   = (bf16*)alloc((size_t)D_STATE * D_MODEL * 2);
  bf16*  WCb   = (bf16*)alloc((size_t)D_STATE * D_MODEL * 2);
  bf16*  Woutb = (bf16*)alloc((size_t)D_MODEL * KBIG * 2);        // 18.9 MB
  float* lam   = (float*)alloc((size_t)D_MODEL * D_STATE * 4);
  float* delta = (float*)alloc((size_t)ROWS * D_MODEL * 4);       // 12.6 MB
  float* Bt    = (float*)alloc((size_t)ROWS * D_STATE * 4);
  float* Ct    = (float*)alloc((size_t)ROWS * D_STATE * 4);
  float* P     = (float*)alloc((size_t)BATCH * D_MODEL * NCHUNK * D_STATE * 4);
  float* Q     = (float*)alloc((size_t)BATCH * D_MODEL * NCHUNK * D_STATE * 4);
  float* Hs    = (float*)alloc((size_t)BATCH * D_MODEL * NCHUNK * D_STATE * 4);
  bf16*  flat  = (bf16*)alloc((size_t)ROWS * KBIG * 2);           // 100.7 MB

  auto cvt = [&](const float* in, bf16* o, int n) {
    cvt_f32_bf16_kernel<<<(n + 255) / 256, 256, 0, stream>>>(in, o, n);
  };
  cvt(x,    xb,    ROWS * D_MODEL);
  cvt(Wd,   Wdb,   D_MODEL * D_MODEL);
  cvt(WB,   WBb,   D_STATE * D_MODEL);
  cvt(WC,   WCb,   D_STATE * D_MODEL);
  cvt(Wout, Woutb, D_MODEL * KBIG);
  lam_kernel<<<(D_MODEL * D_STATE + 255) / 256, 256, 0, stream>>>(llam, lam,
                                                                  D_MODEL * D_STATE);

  // GEMM front-end: delta (4096x768, K=768) + Bt/Ct
  gemm128_kernel<D_MODEL, true><<<192, 256, 0, stream>>>(xb, Wdb, bd, delta);
  btct_kernel<<<32, 256, 0, stream>>>(xb, WBb, WCb, Bt, Ct);

  // Chunked scan (24576-way parallel pass1/pass3, tiny pass2)
  scan_pass1_kernel<<<(BATCH * D_MODEL * NCHUNK) / 256, 256, 0, stream>>>(
      delta, Bt, lam, P, Q);
  scan_pass2_kernel<<<(BATCH * D_MODEL) / 256, 256, 0, stream>>>(P, Q, Hs);
  scan_pass3_kernel<<<(BATCH * D_MODEL * NCHUNK) / 256, 256, 0, stream>>>(
      delta, Bt, Ct, lam, Hs, flat);

  // Dominant GEMM: 4096 x 768 x 12288 (77.3 GFLOP)
  gemm128_kernel<KBIG, false><<<192, 256, 0, stream>>>(flat, Woutb, bout, out);
}
